// PhenotypeNet_67345087201961
// MI455X (gfx1250) — matrix operand order repaired and checked by reading, not compiled, not run on hardware
//
#include <hip/hip_runtime.h>
#include <hip/hip_bf16.h>

// ---------------- problem constants (from reference) ----------------
#define NN   200000   // samples
#define DD   128      // latent dim
#define TT   8        // tasks
#define PPAT 2000     // patients
#define NPH  9        // phenos
#define HH   2048     // hidden
#define TD   1024     // T*D (fingerprint width)

typedef __attribute__((ext_vector_type(16))) _Float16 v16h;
typedef __attribute__((ext_vector_type(8)))  _Float16 v8h;
typedef __attribute__((ext_vector_type(8)))  float    v8f;

union FragA { v16h v; v8h h8[2]; };

// ---------------- init workspace ----------------
__global__ __launch_bounds__(256)
void pn_init_kernel(float* pt_sum, float* pt_cnt, int* first) {
    long long i = (long long)blockIdx.x * blockDim.x + threadIdx.x;
    if (i < (long long)PPAT * TT * DD) pt_sum[i] = 0.f;
    if (i < PPAT * TT)                 pt_cnt[i] = 0.f;
    if (i < PPAT)                      first[i]  = NN;
}

// ---------------- scatter: pt_sum / pt_cnt / first ----------------
__global__ __launch_bounds__(128)
void pn_scatter_kernel(const float* __restrict__ mz,
                       const int* __restrict__ tasks,
                       const int* __restrict__ tmask,
                       const int* __restrict__ pid,
                       const int* __restrict__ pmask,
                       float* __restrict__ pt_sum,
                       float* __restrict__ pt_cnt,
                       int*   __restrict__ first) {
    const int i = blockIdx.x;
    if (tmask[i] != 1 || pmask[i] != 1) return;           // invalid sample: weight 0
    const int p   = pid[i];
    const int seg = p * TT + tasks[i];
    const int d   = threadIdx.x;
    atomicAdd(&pt_sum[(long long)seg * DD + d], mz[(long long)i * DD + d]);
    if (d == 0) {
        atomicAdd(&pt_cnt[seg], 1.0f);
        atomicMin(&first[p], i);
    }
}

// ---------------- per-task fallback mean ----------------
__global__ __launch_bounds__(128)
void pn_taskmean_kernel(const float* __restrict__ pt_sum,
                        const float* __restrict__ pt_cnt,
                        float* __restrict__ task_mean) {
    const int idx = blockIdx.x * blockDim.x + threadIdx.x;   // 0 .. T*D-1
    if (idx >= TT * DD) return;
    const int t = idx / DD;
    const int d = idx % DD;
    float s = 0.f, c = 0.f;
    for (int p = 0; p < PPAT; ++p) {
        s += pt_sum[((long long)p * TT + t) * DD + d];
        c += pt_cnt[p * TT + t];
    }
    task_mean[t * DD + d] = s / fmaxf(c, 1.0f);
}

// ---------------- fingerprint finalize -> f16 A matrix ----------------
__global__ __launch_bounds__(128)
void pn_fingerprint_kernel(const float* __restrict__ pt_sum,
                           const float* __restrict__ pt_cnt,
                           const float* __restrict__ task_mean,
                           _Float16* __restrict__ Ah) {
    const int pt = blockIdx.x;          // p*T + t  (0..P*T-1)
    const int d  = threadIdx.x;
    const int t  = pt % TT;
    const float c = pt_cnt[pt];
    const float v = (c > 0.f) ? (pt_sum[(long long)pt * DD + d] / c)
                              : task_mean[t * DD + d];
    Ah[(long long)pt * DD + d] = (_Float16)v;   // == fingerprint[p][t*D + d]
}

// ---------------- weight convert + transpose: W[K][N] -> Wt[N][K] f16 ----
__global__ __launch_bounds__(256)
void pn_convT_kernel(const float* __restrict__ W, _Float16* __restrict__ Wt,
                     int K, int N) {
    const long long idx = (long long)blockIdx.x * blockDim.x + threadIdx.x;
    if (idx >= (long long)K * N) return;
    const int k = (int)(idx / N);
    const int n = (int)(idx % N);
    Wt[(long long)n * K + k] = (_Float16)W[idx];
}

// ---------------- WMMA GEMM: C = relu(A @ Bt^T + bias), f16 in/out, f32 acc
// A  : [M][K] f16 row-major
// Bt : [N][K] f16 row-major (i.e. transposed weights)
// C  : [M][N] f16 row-major
// block = 256 threads = 8 waves; wave w owns the 16 x 64 strip
//   rows [blockIdx.x*16, +16), cols [blockIdx.y*512 + w*64, +64)
// The 16x32 A tile (shared by all 8 waves) is double-buffered through LDS
// with GLOBAL_LOAD_ASYNC_TO_LDS_B128 (ASYNCcnt); waves read fragments with
// ds_load_b128. B fragments stream straight from L2.
#define AROW_PAD 40   // halves per LDS row (80B) -> conflict-free ds_load_b128
__global__ __launch_bounds__(256)
void pn_gemm_relu_kernel(const _Float16* __restrict__ A,
                         const _Float16* __restrict__ Bt,
                         const float* __restrict__ bias,
                         _Float16* __restrict__ C,
                         int M, int K, int N) {
    __shared__ _Float16 smemA[2][16][AROW_PAD];

    const int tid  = threadIdx.x;
    const int lane = tid & 31;
    const int wave = tid >> 5;
    const int r    = lane & 15;     // A: row within tile / B,C: column within tile
    const int h    = lane >> 4;     // lane half selector

    const int mrow  = blockIdx.x * 16;
    const int nbase = blockIdx.y * 512 + wave * 64;

    v8f acc[4];
#pragma unroll
    for (int t = 0; t < 4; ++t)
        acc[t] = (v8f){0.f, 0.f, 0.f, 0.f, 0.f, 0.f, 0.f, 0.f};

    // async A-tile copy: tid<64 -> lane copies 8 halves (16B):
    //   row = tid>>2 (0..15), chunk = tid&3 (k offset chunk*8)
    const int arow = tid >> 2;
    const int achk = tid & 3;
    const _Float16* Abase = A + (long long)(mrow + arow) * K + achk * 8;

#define PN_ISSUE_A(buf, k0)                                                    \
    do {                                                                       \
        if (tid < 64) {                                                        \
            unsigned long long ga =                                            \
                (unsigned long long)(const void*)(Abase + (k0));               \
            unsigned lo =                                                      \
                (unsigned)(unsigned long long)&smemA[(buf)][arow][achk * 8];   \
            asm volatile("global_load_async_to_lds_b128 %0, %1, off"           \
                         :: "v"(lo), "v"(ga) : "memory");                      \
        }                                                                      \
    } while (0)

    PN_ISSUE_A(0, 0);

    for (int k0 = 0; k0 < K; k0 += 32) {
        const int cur = (k0 >> 5) & 1;
        // copy into buf[cur] (issued last iter / prologue) is done:
        asm volatile("s_wait_asynccnt 0x0" ::: "memory");
        __syncthreads();                       // publish buf[cur] to all waves
        if (k0 + 32 < K) PN_ISSUE_A(cur ^ 1, k0 + 32);  // pipeline next tile

        // A fragment (16x32) from LDS: lane row = r;
        //   halves 0..15 -> K = k0 + h*16 + e
        FragA fa;
        fa.h8[0] = *(const v8h*)&smemA[cur][r][h * 16];
        fa.h8[1] = *(const v8h*)&smemA[cur][r][h * 16 + 8];

#pragma unroll
        for (int t = 0; t < 4; ++t) {
            // B fragment (32x16): lane column = r, K chunk = k0 + h*16 .. +15,
            // contiguous in K because Bt is [N][K].
            const _Float16* Bp = Bt + (long long)(nbase + t * 16 + r) * K + k0 + h * 16;
            FragA fb;
            fb.h8[0] = *(const v8h*)(Bp);
            fb.h8[1] = *(const v8h*)(Bp + 8);
            __builtin_prefetch((const void*)(Bp + 32), 0, 3);  // next K-step, near
            acc[t] = __builtin_amdgcn_wmma_f32_16x16x32_f16(
                /*neg_a=*/false, fa.v, /*neg_b=*/false, fb.v,
                /*c_mod=*/(short)0, acc[t], /*reuse_a=*/false, /*reuse_b=*/false);
        }
    }
#undef PN_ISSUE_A

    // Epilogue: C/D layout — lane column = r, VGPR v holds row v + 8*h.
#pragma unroll
    for (int t = 0; t < 4; ++t) {
        const int ncol = nbase + t * 16 + r;
        const float bv = bias[ncol];
#pragma unroll
        for (int v = 0; v < 8; ++v) {
            const int m = mrow + h * 8 + v;
            if (m < M) {
                float val = acc[t][v] + bv;
                val = val > 0.f ? val : 0.f;
                C[(long long)m * N + ncol] = (_Float16)val;
            }
        }
    }
}

// ---------------- head: latent = h2 @ W3 + b3 ; out = sigmoid(latent) ----
// also gathers uni_phenos = phenos[clip(first[p], 0, N-1)]
// d_out layout: [out (P*NP)] [uni_phenos (P*NP)] [pheno_latent (P*NP)]
__global__ __launch_bounds__(256)
void pn_head_kernel(const _Float16* __restrict__ h2,
                    const float* __restrict__ W3,
                    const float* __restrict__ b3,
                    const float* __restrict__ phenos,
                    const int* __restrict__ first,
                    float* __restrict__ out) {
    const int p   = blockIdx.x;
    const int tid = threadIdx.x;

    float acc[NPH];
#pragma unroll
    for (int j = 0; j < NPH; ++j) acc[j] = 0.f;

    for (int k = tid; k < HH; k += 256) {
        const float hv = (float)h2[(long long)p * HH + k];
#pragma unroll
        for (int j = 0; j < NPH; ++j) acc[j] += hv * W3[k * NPH + j];
    }

    __shared__ float red[NPH][256];
#pragma unroll
    for (int j = 0; j < NPH; ++j) red[j][tid] = acc[j];
    __syncthreads();

    if (tid < NPH) {
        float s = 0.f;
        for (int i = 0; i < 256; ++i) s += red[tid][i];
        const float lat = s + b3[tid];
        out[2 * PPAT * NPH + p * NPH + tid] = lat;                 // pheno_latent
        out[p * NPH + tid] = 1.f / (1.f + __expf(-lat));           // sigmoid
    }
    if (tid >= 32 && tid < 32 + NPH) {
        const int j = tid - 32;
        int idx = first[p];
        if (idx > NN - 1) idx = NN - 1;
        if (idx < 0) idx = 0;
        out[PPAT * NPH + p * NPH + j] = phenos[(long long)idx * NPH + j];
    }
}

// ---------------- launch ----------------
static inline size_t pn_align(size_t x) { return (x + 255) & ~(size_t)255; }

extern "C" void kernel_launch(void* const* d_in, const int* in_sizes, int n_in,
                              void* d_out, int out_size, void* d_ws, size_t ws_size,
                              hipStream_t stream) {
    (void)in_sizes; (void)n_in; (void)out_size; (void)ws_size;

    const float* motion_z  = (const float*)d_in[0];
    const int*   tasks     = (const int*)  d_in[1];
    const int*   tmask     = (const int*)  d_in[2];
    const int*   pids      = (const int*)  d_in[3];
    const float* phenos    = (const float*)d_in[4];
    const int*   pmask     = (const int*)  d_in[5];
    const float* W1        = (const float*)d_in[6];
    const float* b1        = (const float*)d_in[7];
    const float* W2        = (const float*)d_in[8];
    const float* b2        = (const float*)d_in[9];
    const float* W3        = (const float*)d_in[10];
    const float* b3        = (const float*)d_in[11];
    float* out = (float*)d_out;

    // workspace carve-up
    char* w = (char*)d_ws;
    size_t o = 0;
    float*    pt_sum    = (float*)(w + o); o += pn_align((size_t)PPAT * TT * DD * 4);
    float*    pt_cnt    = (float*)(w + o); o += pn_align((size_t)PPAT * TT * 4);
    float*    task_mean = (float*)(w + o); o += pn_align((size_t)TT * DD * 4);
    int*      first     = (int*)  (w + o); o += pn_align((size_t)PPAT * 4);
    _Float16* Ah        = (_Float16*)(w + o); o += pn_align((size_t)PPAT * TD * 2);
    _Float16* W1t       = (_Float16*)(w + o); o += pn_align((size_t)HH * TD * 2);
    _Float16* W2t       = (_Float16*)(w + o); o += pn_align((size_t)HH * HH * 2);
    _Float16* h1        = (_Float16*)(w + o); o += pn_align((size_t)PPAT * HH * 2);
    _Float16* h2        = (_Float16*)(w + o); o += pn_align((size_t)PPAT * HH * 2);

    // 1) init accumulators
    {
        const long long tot = (long long)PPAT * TT * DD;
        pn_init_kernel<<<(unsigned)((tot + 255) / 256), 256, 0, stream>>>(pt_sum, pt_cnt, first);
    }
    // 2) scatter-reduce per (patient, task)
    pn_scatter_kernel<<<NN, DD, 0, stream>>>(motion_z, tasks, tmask, pids, pmask,
                                             pt_sum, pt_cnt, first);
    // 3) per-task fallback mean
    pn_taskmean_kernel<<<(TT * DD + 127) / 128, 128, 0, stream>>>(pt_sum, pt_cnt, task_mean);
    // 4) fingerprint -> f16 A matrix [P][T*D]
    pn_fingerprint_kernel<<<PPAT * TT, DD, 0, stream>>>(pt_sum, pt_cnt, task_mean, Ah);
    // 5) convert+transpose weights to f16 [N][K]
    {
        const long long e1 = (long long)TD * HH;
        pn_convT_kernel<<<(unsigned)((e1 + 255) / 256), 256, 0, stream>>>(W1, W1t, TD, HH);
        const long long e2 = (long long)HH * HH;
        pn_convT_kernel<<<(unsigned)((e2 + 255) / 256), 256, 0, stream>>>(W2, W2t, HH, HH);
    }
    // 6) GEMM1: h1 = relu(Ah[2000,1024] @ W1 + b1)   -> [2000,2048] f16
    {
        dim3 g((PPAT + 15) / 16, (HH + 511) / 512);
        pn_gemm_relu_kernel<<<g, 256, 0, stream>>>(Ah, W1t, b1, h1, PPAT, TD, HH);
    }
    // 7) GEMM2: h2 = relu(h1[2000,2048] @ W2 + b2)   -> [2000,2048] f16
    {
        dim3 g((PPAT + 15) / 16, (HH + 511) / 512);
        pn_gemm_relu_kernel<<<g, 256, 0, stream>>>(h1, W2t, b2, h2, PPAT, HH, HH);
    }
    // 8) head: latent/sigmoid/uni_phenos -> d_out
    pn_head_kernel<<<PPAT, 256, 0, stream>>>(h2, W3, b3, phenos, first, out);
}
